// GraphAttentionLayer_25735444038432
// MI455X (gfx1250) — matrix operand order repaired and checked by reading, not compiled, not run on hardware
//
#include <hip/hip_runtime.h>
#include <hip/hip_bf16.h>

// GAT layer: B=4, N=4096, IN=128, OUT=128, H=4 heads, D=32, leaky slope 0.2
//  kernel 1 (gat_proj): Wh = h @ W^T  (f16 WMMA, f32 accum) -> d_ws [b][head][n][32] f16
//  kernel 2 (gat_attn): flash-attention per (b,head), 64-key tiles streamed into
//                       LDS by the Tensor Data Mover (tensor_load_to_lds, TENSORcnt),
//                       double-buffered; online softmax of leaky_relu(Q K^T); O += P V.

#define BN    4
#define NN    4096
#define INDIM 128
#define OUTD  128
#define NH    4
#define HD    32
#define SLOPE 0.2f
#define KT    64                      // keys per tile
#define NT    (NN / KT)               // 64 tiles

typedef __attribute__((ext_vector_type(16))) _Float16 v16h;
typedef __attribute__((ext_vector_type(8)))  _Float16 v8h;
typedef __attribute__((ext_vector_type(4)))  _Float16 v4h;
typedef __attribute__((ext_vector_type(8)))  float    v8f;
typedef __attribute__((ext_vector_type(4)))  unsigned int u32x4;
typedef __attribute__((ext_vector_type(8)))  int      i32x8;
typedef __attribute__((ext_vector_type(4)))  int      i32x4;

static __device__ inline v16h mk16(v8h a, v8h b) {
  v16h r;
#pragma unroll
  for (int i = 0; i < 8; ++i) { r[i] = a[i]; r[i + 8] = b[i]; }
  return r;
}

static __device__ inline v8f wmma_f16(v16h a, v16h b, v8f c) {
  return __builtin_amdgcn_wmma_f32_16x16x32_f16(
      false, a, false, b, (short)0, c, false, false);
}

// ---- Tensor Data Mover: contiguous copy of 64 x 8B (512 bytes) global -> LDS
static __device__ inline void tdm_copy_512B(const void* gsrc, unsigned lds_off) {
  unsigned long long ga = (unsigned long long)(size_t)gsrc;
  u32x4 g0;
  g0[0] = 1u;                                          // count=1, user mode
  g0[1] = lds_off;                                     // lds_addr
  g0[2] = (unsigned)(ga & 0xFFFFFFFFu);                // global_addr[31:0]
  g0[3] = (unsigned)((ga >> 32) & 0x1FFFFFFu)          // global_addr[56:32]
        | (2u << 30);                                  // type = 2 ("image")
  i32x8 g1;
  g1[0] = (int)(3u << 16);   // workgroup_mask=0 | data_size=3 (8 bytes)
  g1[1] = (int)(64u << 16);  // tensor_dim0[15:0] = 64 (8B units)
  g1[2] = (int)(1u << 16);   // tensor_dim0[31:16]=0 | tensor_dim1[15:0] = 1
  g1[3] = (int)(64u << 16);  // tensor_dim1[31:16]=0 | tile_dim0 = 64
  g1[4] = 1;                 // tile_dim1 = 1 | tile_dim2 = 0
  g1[5] = 64;                // tensor_dim0_stride[31:0] = 64
  g1[6] = 0;
  g1[7] = 0;
#if defined(__clang_major__) && (__clang_major__ >= 23)
  __builtin_amdgcn_tensor_load_to_lds(g0, g1, (i32x4)0, (i32x4)0, (i32x8)0, 0);
#else
  __builtin_amdgcn_tensor_load_to_lds(g0, g1, (i32x4)0, (i32x4)0, 0);
#endif
}

// ---------------------------------------------------------------------------
// Kernel 1: projection  Wh[b][head][n][d] = sum_k h[b][n][k] * W[head*32+d][k]
// 128 threads (4 waves); each wave: 16 rows x 128 cols. grid = (B*N)/64.
// ---------------------------------------------------------------------------
__global__ __launch_bounds__(128) void gat_proj(const float* __restrict__ h,
                                                const float* __restrict__ W,
                                                _Float16* __restrict__ wh) {
  __shared__ __align__(16) _Float16 Wl[INDIM * OUTD];
  __shared__ __align__(16) _Float16 Al[4][16 * INDIM];

  const int tid  = threadIdx.x;
  const int lane = tid & 31;
  const int wave = tid >> 5;
  const int hlf  = (lane >> 4) & 1;
  const int l15  = lane & 15;
  const int r0w  = blockIdx.x * 64 + wave * 16;

  for (int i = tid; i < (INDIM * OUTD) / 4; i += 128) {
    float4 w4 = ((const float4*)W)[i];
    v4h v = {(_Float16)w4.x, (_Float16)w4.y, (_Float16)w4.z, (_Float16)w4.w};
    *(v4h*)&Wl[i * 4] = v;
  }
  for (int i = lane; i < (16 * INDIM) / 4; i += 32) {
    const int row = i >> 5, c4 = i & 31;
    float4 a4 = ((const float4*)(h + (size_t)(r0w + row) * INDIM))[c4];
    v4h v = {(_Float16)a4.x, (_Float16)a4.y, (_Float16)a4.z, (_Float16)a4.w};
    *(v4h*)&Al[wave][row * INDIM + c4 * 4] = v;
  }
  __syncthreads();

  const _Float16* arow = &Al[wave][l15 * INDIM];
  v16h af[4];
#pragma unroll
  for (int kk = 0; kk < 4; ++kk)
    af[kk] = mk16(*(const v8h*)(arow + kk * 32 + hlf * 8),
                  *(const v8h*)(arow + kk * 32 + 16 + hlf * 8));

#pragma unroll
  for (int ot = 0; ot < 8; ++ot) {
    v8f acc = {};
#pragma unroll
    for (int kk = 0; kk < 4; ++kk) {
      const _Float16* wrow = &Wl[(ot * 16 + l15) * INDIM + kk * 32 + hlf * 8];
      v16h bf = mk16(*(const v8h*)wrow, *(const v8h*)(wrow + 16));
      acc = wmma_f16(af[kk], bf, acc);
    }
    const int o = ot * 16 + l15;
#pragma unroll
    for (int v = 0; v < 8; ++v) {
      const int m = r0w + v + 8 * hlf;
      const int b = m >> 12, n = m & (NN - 1);
      wh[((size_t)(b * NH + (o >> 5)) * NN + n) * HD + (o & 31)] = (_Float16)acc[v];
    }
  }
}

// ---------------------------------------------------------------------------
// Kernel 2: flash attention per (b, head). 256 threads = 8 waves; wave owns
// 16 query rows; 64-key tiles, TDM double-buffered. grid = (N/128, H, B).
// ---------------------------------------------------------------------------
__global__ __launch_bounds__(256) void gat_attn(const _Float16* __restrict__ wh,
                                                float* __restrict__ out) {
  __shared__ __align__(16) _Float16 kv [2][KT * HD];    // [key][d], double buffered (8 KB)
  __shared__ __align__(16) _Float16 kvT[HD * KT];       // [d][key] for V fragments (4 KB)
  __shared__ __align__(16) _Float16 Ps [8][16 * KT];    // per-wave P scratch (16 KB)

  const int tid  = threadIdx.x;
  const int lane = tid & 31;
  const int wave = tid >> 5;
  const int hlf  = (lane >> 4) & 1;
  const int l15  = lane & 15;
  const int head = blockIdx.y;
  const int b    = blockIdx.z;
  const int q0   = blockIdx.x * 128 + wave * 16;

  const _Float16* base = wh + (size_t)(b * NH + head) * NN * HD;

  // Persistent Q A-fragment: row = q0+l15, d halves split by lane-half
  const _Float16* qrow = base + (size_t)(q0 + l15) * HD;
  const v16h qf = mk16(*(const v8h*)(qrow + (hlf ? 8 : 0)),
                       *(const v8h*)(qrow + (hlf ? 24 : 16)));

  v8f o0 = {}, o1 = {};
  float mrow[8], lrow[8];
#pragma unroll
  for (int i = 0; i < 8; ++i) { mrow[i] = -3.0e38f; lrow[i] = 0.0f; }

  // Prologue: each wave DMAs its own 8-key (512B) slice of tile 0
  tdm_copy_512B(base + (size_t)wave * 8 * HD,
                (unsigned)(size_t)&kv[0][wave * 8 * HD]);

  for (int kt = 0; kt < NT; ++kt) {
    const int cur = kt & 1;
    __builtin_amdgcn_s_wait_tensorcnt(0);   // my slice of tile kt landed in LDS
    __syncthreads();                        // all 8 slices visible

    // Kick DMA of next tile into the other buffer (overlaps compute below)
    if (kt + 1 < NT)
      tdm_copy_512B(base + ((size_t)(kt + 1) * KT + wave * 8) * HD,
                    (unsigned)(size_t)&kv[cur ^ 1][wave * 8 * HD]);

    // Build transposed copy kvT[d][key] from kv[cur] (for V B-fragments)
    {
      const int key = tid >> 2;             // 0..63
      const int d0  = (tid & 3) * 8;        // 0,8,16,24
      v8h t = *(const v8h*)&kv[cur][key * HD + d0];
#pragma unroll
      for (int j = 0; j < 8; ++j) kvT[(d0 + j) * KT + key] = t[j];
    }
    __syncthreads();                        // kvT ready

    // --- S = Q K^T for 64 keys: four 16x16 WMMAs (K-dim = d = 32)
    v8f s[4];
#pragma unroll
    for (int j = 0; j < 4; ++j) {
      const _Float16* kp = &kv[cur][(j * 16 + l15) * HD + hlf * 8];
      v16h kb = mk16(*(const v8h*)kp, *(const v8h*)(kp + 16));
      v8f z = {};
      s[j] = wmma_f16(qf, kb, z);
    }

    // --- leaky relu
#pragma unroll
    for (int j = 0; j < 4; ++j)
#pragma unroll
      for (int i = 0; i < 8; ++i)
        s[j][i] = s[j][i] > 0.0f ? s[j][i] : SLOPE * s[j][i];

    // --- online softmax over 64 columns
    float t[8];
#pragma unroll
    for (int i = 0; i < 8; ++i)
      t[i] = fmaxf(fmaxf(s[0][i], s[1][i]), fmaxf(s[2][i], s[3][i]));
#pragma unroll
    for (int x = 1; x <= 8; x <<= 1)
#pragma unroll
      for (int i = 0; i < 8; ++i) t[i] = fmaxf(t[i], __shfl_xor(t[i], x));

    float alpha[8], rs[8];
    float p[4][8];
#pragma unroll
    for (int i = 0; i < 8; ++i) {
      const float mn = fmaxf(mrow[i], t[i]);
      alpha[i] = __expf(mrow[i] - mn);
      mrow[i]  = mn;
      rs[i] = 0.0f;
#pragma unroll
      for (int j = 0; j < 4; ++j) {
        p[j][i] = __expf(s[j][i] - mn);
        rs[i] += p[j][i];
      }
    }
#pragma unroll
    for (int x = 1; x <= 8; x <<= 1)
#pragma unroll
      for (int i = 0; i < 8; ++i) rs[i] += __shfl_xor(rs[i], x);
#pragma unroll
    for (int i = 0; i < 8; ++i) {
      lrow[i] = lrow[i] * alpha[i] + rs[i];
      o0[i] *= alpha[i];
      o1[i] *= alpha[i];
    }

    // --- re-layout P: C-frag -> A-frag via per-wave LDS scratch (f16)
    _Float16* pb = &Ps[wave][0];
#pragma unroll
    for (int j = 0; j < 4; ++j)
#pragma unroll
      for (int v = 0; v < 8; ++v)
        pb[(v + 8 * hlf) * KT + j * 16 + l15] = (_Float16)p[j][v];

    const _Float16* pr = pb + l15 * KT;
    const v16h pf0 = mk16(*(const v8h*)(pr + hlf * 16),
                          *(const v8h*)(pr + hlf * 16 + 8));
    const v16h pf1 = mk16(*(const v8h*)(pr + 32 + hlf * 16),
                          *(const v8h*)(pr + 40 + hlf * 16));

    // --- O += P V: B[m][d] = kvT[d][m], key groups g=0 (m 0..31), g=1 (m 32..63)
    const v16h vb00 = mk16(*(const v8h*)&kvT[l15 * KT + hlf * 16],
                           *(const v8h*)&kvT[l15 * KT + hlf * 16 + 8]);
    const v16h vb10 = mk16(*(const v8h*)&kvT[l15 * KT + 32 + hlf * 16],
                           *(const v8h*)&kvT[l15 * KT + 32 + hlf * 16 + 8]);
    const v16h vb01 = mk16(*(const v8h*)&kvT[(16 + l15) * KT + hlf * 16],
                           *(const v8h*)&kvT[(16 + l15) * KT + hlf * 16 + 8]);
    const v16h vb11 = mk16(*(const v8h*)&kvT[(16 + l15) * KT + 32 + hlf * 16],
                           *(const v8h*)&kvT[(16 + l15) * KT + 32 + hlf * 16 + 8]);
    o0 = wmma_f16(pf0, vb00, o0);
    o0 = wmma_f16(pf1, vb10, o0);
    o1 = wmma_f16(pf0, vb01, o1);
    o1 = wmma_f16(pf1, vb11, o1);
  }

  // --- normalize and write out[b][n][head*32 + d]
#pragma unroll
  for (int v = 0; v < 8; ++v) {
    const float inv = 1.0f / lrow[v];
    const int n = q0 + v + 8 * hlf;
    float* orow = out + ((size_t)b * NN + n) * OUTD + head * HD;
    orow[l15]      = o0[v] * inv;
    orow[16 + l15] = o1[v] * inv;
  }
}

extern "C" void kernel_launch(void* const* d_in, const int* in_sizes, int n_in,
                              void* d_out, int out_size, void* d_ws, size_t ws_size,
                              hipStream_t stream) {
  const float* h = (const float*)d_in[0];   // [B, N, 128] f32
  const float* W = (const float*)d_in[1];   // [128, 128]  f32
  float* out = (float*)d_out;               // [B, N, 128] f32
  _Float16* wh = (_Float16*)d_ws;           // [B, H, N, 32] f16 (4 MiB)

  gat_proj<<<dim3((BN * NN) / 64), 128, 0, stream>>>(h, W, wh);
  gat_attn<<<dim3(NN / 128, NH, BN), 256, 0, stream>>>(wh, out);
}
